// MLLTRSCLloss_32547262169410
// MI455X (gfx1250) — matrix operand order repaired and checked by reading, not compiled, not run on hardware
//
#include <hip/hip_runtime.h>
#include <hip/hip_bf16.h>

typedef __attribute__((ext_vector_type(8)))  _Float16 v8h;
typedef __attribute__((ext_vector_type(16))) _Float16 v16h;
typedef __attribute__((ext_vector_type(8)))  float    v8f;

static constexpr int D_DIM = 768;
static constexpr int C_DIM = 97;
static constexpr float TAU = 2.0f;

// ---------------------------------------------------------------------------
// Kernel 1: zero the r/p/c partial arrays (3*N floats, contiguous)
// ---------------------------------------------------------------------------
__global__ void zero_kernel(float* __restrict__ buf, int n) {
    int i = blockIdx.x * blockDim.x + threadIdx.x;
    if (i < n) buf[i] = 0.0f;
}

// ---------------------------------------------------------------------------
// Kernel 2: pack multi-hot labels (N x 97 int32, values 0/1) into 128-bit
// masks, and compute label_mask[i] = (labels[i,0] != 1)
// ---------------------------------------------------------------------------
__global__ void pack_kernel(const int* __restrict__ labels,
                            uint4* __restrict__ bits,
                            float* __restrict__ lmask, int N) {
    int i = blockIdx.x * blockDim.x + threadIdx.x;
    if (i >= N) return;
    const int* row = labels + (size_t)i * C_DIM;
    unsigned w0 = 0, w1 = 0, w2 = 0, w3 = 0;
    #pragma unroll 4
    for (int c = 0; c < C_DIM; ++c) {
        unsigned b = (row[c] != 0) ? 1u : 0u;
        unsigned bit = b << (c & 31);
        int word = c >> 5;
        if (word == 0) w0 |= bit;
        else if (word == 1) w1 |= bit;
        else if (word == 2) w2 |= bit;
        else w3 |= bit;
    }
    bits[i] = make_uint4(w0, w1, w2, w3);
    lmask[i] = (row[0] != 1) ? 1.0f : 0.0f;
}

// ---------------------------------------------------------------------------
// Kernel 3: convert features to f16 and compute m_i = (sum_k h_ik^2)/TAU,
// which equals the row max of sims (diagonal dominates by Cauchy-Schwarz).
// ---------------------------------------------------------------------------
__global__ __launch_bounds__(256) void cvt_kernel(const float* __restrict__ feats,
                                                  _Float16* __restrict__ f16,
                                                  float* __restrict__ m) {
    const int row = blockIdx.x;
    const int tid = threadIdx.x;
    const float* src = feats + (size_t)row * D_DIM;
    _Float16* dst = f16 + (size_t)row * D_DIM;
    float ss = 0.0f;
    for (int k = tid; k < D_DIM; k += 256) {
        float x = src[k];
        _Float16 h = (_Float16)x;
        dst[k] = h;
        float hf = (float)h;
        ss += hf * hf;
    }
    #pragma unroll
    for (int off = 16; off >= 1; off >>= 1) ss += __shfl_xor(ss, off, 32);
    __shared__ float red[8];
    int wave = tid >> 5, lane = tid & 31;
    if (lane == 0) red[wave] = ss;
    __syncthreads();
    if (tid == 0) {
        float tot = 0.0f;
        #pragma unroll
        for (int w = 0; w < 8; ++w) tot += red[w];
        m[row] = tot * (1.0f / TAU);
    }
}

// ---------------------------------------------------------------------------
// Fragment helpers (CDNA5 ISA per-lane layouts)
// A (16x32 f16): lane l -> row M=l%16; lanes<16 hold K {k0+0..7, k0+16..23},
//                lanes>=16 hold K {k0+8..15, k0+24..31}
// B (32x16 f16): lane l -> col N=l%16; lanes<16 hold K {k0+0..7, k0+16..23},
//                lanes>=16 hold K {k0+8..15, k0+24..31}
// ---------------------------------------------------------------------------
__device__ inline v16h load_fragA(const _Float16* __restrict__ rowbase, int k0, int upper) {
    int koff = k0 + (upper ? 8 : 0);
    v8h lo = *(const v8h*)(rowbase + koff);
    v8h hi = *(const v8h*)(rowbase + koff + 16);
    return __builtin_shufflevector(lo, hi, 0,1,2,3,4,5,6,7,8,9,10,11,12,13,14,15);
}

__device__ inline v16h load_fragB_lds(const _Float16* ldsrow, int upper) {
    int koff = (upper ? 8 : 0);
    v8h lo = *(const v8h*)(ldsrow + koff);
    v8h hi = *(const v8h*)(ldsrow + koff + 16);
    return __builtin_shufflevector(lo, hi, 0,1,2,3,4,5,6,7,8,9,10,11,12,13,14,15);
}

// CDNA5 async copy: 16 bytes global -> LDS, tracked by ASYNCcnt (GV mode).
__device__ inline void async_copy_b128(unsigned lds_byte_off, const void* gptr) {
    asm volatile("global_load_async_to_lds_b128 %0, %1, off"
                 :: "v"(lds_byte_off),
                    "v"((unsigned long long)(uintptr_t)gptr)
                 : "memory");
}

// ---------------------------------------------------------------------------
// Kernel 4: fused WMMA similarity tile + masked softmax statistics.
// Grid: (N/128, N/256) blocks of 256 threads (8 waves).
// Workgroup tile: 256 i-rows x 128 j-cols. Each wave owns 32 i-rows
// (two A fragments, 16 f32 accumulators), so every B fragment staged in
// LDS feeds two WMMAs (1 ds_load_b128 per WMMA). The 128x64-half B panel
// per K super-step is staged with async-to-LDS copies, double buffered:
// issue next / s_wait_asynccnt / barrier / 32 x WMMA / barrier.
// ---------------------------------------------------------------------------
__global__ __launch_bounds__(256) void sim_stats_kernel(
    const _Float16* __restrict__ f16,
    const uint4* __restrict__ bits,
    const float* __restrict__ m,
    float* __restrict__ rPart,
    float* __restrict__ pPart,
    float* __restrict__ cPart,
    int N)
{
    __shared__ _Float16 sB[2][128 * 64];   // 2 x 16 KB: j-panel, K super-step of 64
    __shared__ uint4 sBits[256];
    __shared__ float sM[256];

    const int bi = blockIdx.y, bj = blockIdx.x;
    const int tid = threadIdx.x;
    const int wave = tid >> 5;
    const int lane = tid & 31;
    const int i0 = bi * 256, j0 = bj * 128;

    sBits[tid] = bits[i0 + tid];
    sM[tid]    = m[i0 + tid];

    const int upper = (lane >= 16) ? 1 : 0;
    const int lr = lane & 15;

    const _Float16* aRow0 = f16 + (size_t)(i0 + wave * 32 + lr) * D_DIM;
    const _Float16* aRow1 = aRow0 + (size_t)16 * D_DIM;
    const _Float16* jPanel = f16 + (size_t)j0 * D_DIM;

    const unsigned sbBase = (unsigned)(uintptr_t)(&sB[0][0]);

    // Each thread stages 4 x 16B chunks per super-step (128 rows x 128 bytes).
    const int chunk0 = tid * 4;                 // of 1024 chunks
    // chunk c: row = c>>3, sub = c&7  (8 x 16B per row)

    auto issue_panel = [&](int bufsel, int k0) {
        #pragma unroll
        for (int q = 0; q < 4; ++q) {
            const int c = chunk0 + q;
            const int row = c >> 3;
            const int sub = c & 7;
            const _Float16* g = jPanel + (size_t)row * D_DIM + k0 + sub * 8;
            const unsigned l = sbBase + (unsigned)(bufsel * 128 * 64 * 2 + row * 128 + sub * 16);
            async_copy_b128(l, g);
        }
    };

    issue_panel(0, 0);          // prologue: stage first super-step

    v8f acc0[8], acc1[8];
    v8f zero = {0.f, 0.f, 0.f, 0.f, 0.f, 0.f, 0.f, 0.f};
    #pragma unroll
    for (int t = 0; t < 8; ++t) { acc0[t] = zero; acc1[t] = zero; }

    for (int k0 = 0; k0 < D_DIM; k0 += 64) {
        const int cur = (k0 >> 6) & 1;
        if (k0 + 64 < D_DIM) {
            issue_panel(cur ^ 1, k0 + 64);
            asm volatile("s_wait_asynccnt 0x4" ::: "memory");  // current panel landed
        } else {
            asm volatile("s_wait_asynccnt 0x0" ::: "memory");
        }
        __syncthreads();        // all waves' chunks of the current panel visible

        #pragma unroll
        for (int kk = 0; kk < 64; kk += 32) {
            v16h a0 = load_fragA(aRow0, k0 + kk, upper);
            v16h a1 = load_fragA(aRow1, k0 + kk, upper);
            #pragma unroll
            for (int t = 0; t < 8; ++t) {
                const _Float16* ldsrow = &sB[cur][(t * 16 + lr) * 64 + kk];
                v16h b = load_fragB_lds(ldsrow, upper);
                acc0[t] = __builtin_amdgcn_wmma_f32_16x16x32_f16(
                    false, a0, false, b, (short)0, acc0[t], false, false);
                acc1[t] = __builtin_amdgcn_wmma_f32_16x16x32_f16(
                    false, a1, false, b, (short)0, acc1[t], false, false);
            }
        }
        __syncthreads();        // reads of this buffer done before it is re-filled
    }

    // ---- epilogue: per-element exp + mask stats ----
    // C/D layout: vgpr v, lanes 0-15 -> (M=v, N=lane); lanes 16-31 -> (M=v+8, N=lane-16)
    #pragma unroll
    for (int s = 0; s < 2; ++s) {
        float rA[8], pA[8], cA[8];
        #pragma unroll
        for (int v = 0; v < 8; ++v) { rA[v] = 0.f; pA[v] = 0.f; cA[v] = 0.f; }

        #pragma unroll
        for (int t = 0; t < 8; ++t) {
            const int jg = j0 + t * 16 + lr;
            const uint4 bb = bits[jg];
            #pragma unroll
            for (int v = 0; v < 8; ++v) {
                const int iloc = wave * 32 + s * 16 + v + (upper ? 8 : 0);
                const int ig = i0 + iloc;
                const float sv = (s == 0 ? acc0[t][v] : acc1[t][v]) * (1.0f / TAU);
                const uint4 ab = sBits[iloc];
                const bool any = (((ab.x & bb.x) | (ab.y & bb.y) |
                                   (ab.z & bb.z) | (ab.w & bb.w)) != 0u);
                const bool diag = (ig == jg);
                const float e = __expf(sv - sM[iloc]);
                if (!diag) rA[v] += e;
                if (any && !diag) { pA[v] += sv; cA[v] += 1.0f; }
            }
        }

        #pragma unroll
        for (int v = 0; v < 8; ++v) {
            float r = rA[v], p = pA[v], c = cA[v];
            #pragma unroll
            for (int off = 8; off >= 1; off >>= 1) {
                r += __shfl_xor(r, off, 32);
                p += __shfl_xor(p, off, 32);
                c += __shfl_xor(c, off, 32);
            }
            if (lr == 0) {
                const int ig = i0 + wave * 32 + s * 16 + v + (upper ? 8 : 0);
                atomicAdd(&rPart[ig], r);
                atomicAdd(&pPart[ig], p);
                atomicAdd(&cPart[ig], c);
            }
        }
    }
}

// ---------------------------------------------------------------------------
// Kernel 5: finalize. loss = -TAU * (S1/N + S2*S3/N^2)
// ---------------------------------------------------------------------------
__global__ __launch_bounds__(256) void finalize_kernel(
    const float* __restrict__ rPart,
    const float* __restrict__ pPart,
    const float* __restrict__ cPart,
    const float* __restrict__ m,
    const float* __restrict__ lmask,
    float* __restrict__ out, int N)
{
    float s1 = 0.f, s2 = 0.f, s3 = 0.f;
    for (int i = threadIdx.x; i < N; i += 256) {
        const float logr = __logf(rPart[i]);
        const float ci = cPart[i];
        const float lp1 = (pPart[i] - ci * (m[i] + logr)) * (1.0f / fmaxf(ci, 1.0f));
        const float lp2 = (ci == 0.0f) ? -logr : 0.0f;
        const float lm = lmask[i];
        s1 += lp1 * lm;
        s2 += lp2;
        s3 += lm;
    }
    #pragma unroll
    for (int off = 16; off >= 1; off >>= 1) {
        s1 += __shfl_xor(s1, off, 32);
        s2 += __shfl_xor(s2, off, 32);
        s3 += __shfl_xor(s3, off, 32);
    }
    __shared__ float red[3][8];
    const int wave = threadIdx.x >> 5, lane = threadIdx.x & 31;
    if (lane == 0) { red[0][wave] = s1; red[1][wave] = s2; red[2][wave] = s3; }
    __syncthreads();
    if (threadIdx.x == 0) {
        float S1 = 0.f, S2 = 0.f, S3 = 0.f;
        #pragma unroll
        for (int w = 0; w < 8; ++w) { S1 += red[0][w]; S2 += red[1][w]; S3 += red[2][w]; }
        const float Nf = (float)N;
        out[0] = -TAU * (S1 / Nf + S2 * S3 / (Nf * Nf));
    }
}

// ---------------------------------------------------------------------------
extern "C" void kernel_launch(void* const* d_in, const int* in_sizes, int n_in,
                              void* d_out, int out_size, void* d_ws, size_t ws_size,
                              hipStream_t stream) {
    const float* feats = (const float*)d_in[0];
    const int* labels = (const int*)d_in[1];
    const int N = in_sizes[0] / D_DIM;   // 4096

    char* ws = (char*)d_ws;
    size_t off = 0;
    _Float16* f16 = (_Float16*)(ws + off); off += (size_t)N * D_DIM * sizeof(_Float16);
    uint4* bits = (uint4*)(ws + off);      off += (size_t)N * sizeof(uint4);
    float* m = (float*)(ws + off);         off += (size_t)N * sizeof(float);
    float* lmask = (float*)(ws + off);     off += (size_t)N * sizeof(float);
    float* rPart = (float*)(ws + off);     off += (size_t)N * sizeof(float);
    float* pPart = (float*)(ws + off);     off += (size_t)N * sizeof(float);
    float* cPart = (float*)(ws + off);     off += (size_t)N * sizeof(float);

    zero_kernel<<<(3 * N + 255) / 256, 256, 0, stream>>>(rPart, 3 * N);
    pack_kernel<<<(N + 255) / 256, 256, 0, stream>>>(labels, bits, lmask, N);
    cvt_kernel<<<N, 256, 0, stream>>>(feats, f16, m);

    dim3 grid(N / 128, N / 256);
    sim_stats_kernel<<<grid, 256, 0, stream>>>(f16, bits, m, rPart, pPart, cPart, N);

    finalize_kernel<<<1, 256, 0, stream>>>(rPart, pPart, cPart, m, lmask, (float*)d_out, N);
}
